// StreamingTransformerEncoderLayer_146028888392
// MI455X (gfx1250) — compile-verified, matrix operand order checked by reading, not run on hardware
//
#include <hip/hip_runtime.h>
#include <hip/hip_bf16.h>

// ---------------------------------------------------------------------------
// StreamingTransformerEncoderLayer decode step, MI455X (gfx1250, wave32).
//
// Roofline: dominant cost is the KV-cache append (2.1 GB read + 2.1 GB write)
// plus one streaming re-read for attention (2.1 GB) -> ~6.3 GB @ 23.3 TB/s.
// The naive 1.1 TFLOP cache re-projection is algebraically folded into the
// (single-token) query / attention-weighted value, leaving small dense GEMMs
// on v_wmma_f32_16x16x32_bf16 (fp32 accumulate). The cache append itself is
// driven by the Tensor Data Mover (tensor_load_to_lds / tensor_store_from_lds
// + s_wait_tensorcnt), which streams HBM without burning VALU issue slots.
// ---------------------------------------------------------------------------

typedef __attribute__((ext_vector_type(16))) __bf16        v16bf;
typedef __attribute__((ext_vector_type(8)))  float         v8f;
typedef __attribute__((ext_vector_type(8)))  unsigned int  v8u;
typedef __attribute__((ext_vector_type(4)))  unsigned int  u32x4;
typedef __attribute__((ext_vector_type(8)))  int           i32x8;
typedef __attribute__((ext_vector_type(4)))  int           i32x4;

#define D_MODEL 1024
#define NHEAD   16
#define HEAD_DIM 64
#define DFF     4096
#define BATCH   64
#define SEQ     4096          // 4095 cached + 1 new
#define LDS_STRIDE 36         // 16x32 bf16 tile rows, padded (72B) vs 64 banks

__device__ __forceinline__ unsigned short f2bf(float f) {
    unsigned int u = __float_as_uint(f);
    unsigned int r = u + 0x7FFFu + ((u >> 16) & 1u);   // round-nearest-even
    return (unsigned short)(r >> 16);
}
__device__ __forceinline__ unsigned int pack2(float a, float b) {
    return (unsigned int)f2bf(a) | ((unsigned int)f2bf(b) << 16);
}

// ---------------------------------------------------------------------------
// Generic block tile: C[16 x 64] = alpha * A . B (+bias) (+relu)
//   A: fp32 row-major [M][K], lda
//   B_KN=false: B stored [N][K] (weights / comb_k rows)  -> C = A . B^T
//   B_KN=true : B stored [K][N]                          -> C = A . B
// 128 threads = 4 waves; wave w owns the N-subtile [n0+16w, n0+16w+16).
// ---------------------------------------------------------------------------
template <bool B_KN, bool RELU>
__device__ void wmma_gemm_block(const float* __restrict__ A, int lda,
                                const float* __restrict__ B, int ldb,
                                float* __restrict__ C, int ldc,
                                int K, int m0, int n0, float alpha,
                                const float* __restrict__ bias)
{
    __shared__ unsigned short As[16 * LDS_STRIDE];
    __shared__ unsigned short Bs[4 * 16 * LDS_STRIDE];

    const int tid     = threadIdx.x;
    const int wave    = tid >> 5;
    const int lane    = tid & 31;
    const int ln_half = lane >> 4;   // which 16-lane half
    const int r       = lane & 15;   // row (A) / col (B) index within tile
    const int n0w     = n0 + wave * 16;
    unsigned short* BsW = Bs + wave * 16 * LDS_STRIDE;

    v8f acc = {};

    for (int k0 = 0; k0 < K; k0 += 32) {
        __syncthreads();
        // ---- stage A tile (16 x 32), shared by all 4 waves ----
        {
            const int e = tid * 4;            // 512 elems / 128 threads
            const int m = e >> 5;
            const int k = e & 31;
            const float4 f = *(const float4*)(A + (long)(m0 + m) * lda + k0 + k);
            *(uint2*)(&As[m * LDS_STRIDE + k]) =
                make_uint2(pack2(f.x, f.y), pack2(f.z, f.w));
        }
        // ---- stage B tile (16 n x 32 k) for this wave ----
        if (!B_KN) {
            const int n  = lane >> 1;
            const int kb = (lane & 1) << 4;   // 0 or 16
            const float* brow = B + (long)(n0w + n) * ldb + k0 + kb;
            const float4 f0 = *(const float4*)(brow + 0);
            const float4 f1 = *(const float4*)(brow + 4);
            const float4 f2 = *(const float4*)(brow + 8);
            const float4 f3 = *(const float4*)(brow + 12);
            uint2* dst = (uint2*)(BsW + n * LDS_STRIDE + kb);
            dst[0] = make_uint2(pack2(f0.x, f0.y), pack2(f0.z, f0.w));
            dst[1] = make_uint2(pack2(f1.x, f1.y), pack2(f1.z, f1.w));
            dst[2] = make_uint2(pack2(f2.x, f2.y), pack2(f2.z, f2.w));
            dst[3] = make_uint2(pack2(f3.x, f3.y), pack2(f3.z, f3.w));
        } else {
            const int k = lane;               // 0..31
            const float* brow = B + (long)(k0 + k) * ldb + n0w;
            const float4 f0 = *(const float4*)(brow + 0);
            const float4 f1 = *(const float4*)(brow + 4);
            const float4 f2 = *(const float4*)(brow + 8);
            const float4 f3 = *(const float4*)(brow + 12);
            float t[16] = { f0.x, f0.y, f0.z, f0.w, f1.x, f1.y, f1.z, f1.w,
                            f2.x, f2.y, f2.z, f2.w, f3.x, f3.y, f3.z, f3.w };
            #pragma unroll
            for (int n = 0; n < 16; ++n)
                BsW[n * LDS_STRIDE + k] = f2bf(t[n]);
        }
        // prefetch next A slab (emits global_prefetch_b8)
        if (k0 + 32 < K)
            __builtin_prefetch((const void*)(A + (long)(m0 + (tid & 15)) * lda + k0 + 32), 0, 1);
        __syncthreads();

        // ---- assemble fragments: packed bf16x2 per documented layout ----
        v8u au, bu;
        #pragma unroll
        for (int i = 0; i < 8; ++i) {
            const int kb = ((i & 4) ? 16 : 0) + 2 * (i & 3) + 8 * ln_half;
            au[i] = *(const unsigned int*)(&As [r * LDS_STRIDE + kb]);
            bu[i] = *(const unsigned int*)(&BsW[r * LDS_STRIDE + kb]);
        }
        acc = __builtin_amdgcn_wmma_f32_16x16x32_bf16(
                  false, __builtin_bit_cast(v16bf, au),
                  false, __builtin_bit_cast(v16bf, bu),
                  (short)0, acc, false, false);
    }

    // ---- epilogue: C/D layout -> element (M = v + 8*half, N = lane%16) ----
    #pragma unroll
    for (int v = 0; v < 8; ++v) {
        const int row = m0 + v + 8 * ln_half;
        const int col = n0w + r;
        float val = acc[v] * alpha;
        if (bias) val += bias[col];
        if (RELU) val = fmaxf(val, 0.0f);
        C[(long)row * ldc + col] = val;
    }
}

// ---------------------------------------------------------------------------
// GEMM wrapper kernels
// ---------------------------------------------------------------------------
__global__ void __launch_bounds__(128)
k_gemm_qproj(const float* __restrict__ x, const float* __restrict__ wq,
             const float* __restrict__ bq, float* __restrict__ q) {
    // q(64x1024) = x(64x1024) . wq^T + bq     (wq stored [N][K])
    wmma_gemm_block<false, false>(x, D_MODEL, wq, D_MODEL, q, D_MODEL,
                                  D_MODEL, blockIdx.y * 16, blockIdx.x * 64,
                                  1.0f, bq);
}

__global__ void __launch_bounds__(128)
k_gemm_qhat(const float* __restrict__ q, const float* __restrict__ wk,
            float* __restrict__ qhat) {
    // per head h: qhat_h(64b x 1024) = (1/8) * q[:,64h:64h+64] . wk_h   (wk [K][N])
    const int h = blockIdx.z;
    wmma_gemm_block<true, false>(q + h * HEAD_DIM, D_MODEL,
                                 wk + (size_t)h * HEAD_DIM * D_MODEL, D_MODEL,
                                 qhat + h * D_MODEL, NHEAD * D_MODEL,
                                 HEAD_DIM, blockIdx.y * 16, blockIdx.x * 64,
                                 0.125f /* 1/sqrt(64) */, nullptr);
}

__global__ void __launch_bounds__(128)
k_gemm_scores(const float* __restrict__ qhat, const float* __restrict__ combk,
              float* __restrict__ scores) {
    // per batch b: scores(16h x 4096s) = qhat_b(16x1024) . combk_b^T   (combk [N=s][K=d])
    const int b = blockIdx.z;
    wmma_gemm_block<false, false>(qhat + (size_t)b * NHEAD * D_MODEL, D_MODEL,
                                  combk + (size_t)b * SEQ * D_MODEL, D_MODEL,
                                  scores + (size_t)b * NHEAD * SEQ, SEQ,
                                  D_MODEL, 0, blockIdx.x * 64, 1.0f, nullptr);
}

__global__ void __launch_bounds__(128)
k_gemm_vbar(const float* __restrict__ attn, const float* __restrict__ combv,
            float* __restrict__ vbar) {
    // per batch b: vbar(16h x 1024d) = attn_b(16x4096) . combv_b   (combv [K=s][N=d])
    const int b = blockIdx.z;
    wmma_gemm_block<true, false>(attn + (size_t)b * NHEAD * SEQ, SEQ,
                                 combv + (size_t)b * SEQ * D_MODEL, D_MODEL,
                                 vbar + (size_t)b * NHEAD * D_MODEL, D_MODEL,
                                 SEQ, 0, blockIdx.x * 64, 1.0f, nullptr);
}

__global__ void __launch_bounds__(128)
k_gemm_ctx(const float* __restrict__ vbar, const float* __restrict__ wv,
           const float* __restrict__ bv, float* __restrict__ ctx) {
    // per head h: ctx[:, 64h:64h+64] = vbar_h(64b x 1024) . wv_h^T + bv_h  (wv [N][K])
    const int h = blockIdx.z;
    wmma_gemm_block<false, false>(vbar + h * D_MODEL, NHEAD * D_MODEL,
                                  wv + (size_t)h * HEAD_DIM * D_MODEL, D_MODEL,
                                  ctx + h * HEAD_DIM, D_MODEL,
                                  D_MODEL, blockIdx.y * 16, 0, 1.0f,
                                  bv + h * HEAD_DIM);
}

__global__ void __launch_bounds__(128)
k_gemm_outproj(const float* __restrict__ ctx, const float* __restrict__ wo,
               const float* __restrict__ bo, float* __restrict__ aout) {
    wmma_gemm_block<false, false>(ctx, D_MODEL, wo, D_MODEL, aout, D_MODEL,
                                  D_MODEL, blockIdx.y * 16, blockIdx.x * 64,
                                  1.0f, bo);
}

__global__ void __launch_bounds__(128)
k_gemm_ffn1(const float* __restrict__ h, const float* __restrict__ w1,
            const float* __restrict__ b1, float* __restrict__ ff1) {
    wmma_gemm_block<false, true>(h, D_MODEL, w1, D_MODEL, ff1, DFF,
                                 D_MODEL, blockIdx.y * 16, blockIdx.x * 64,
                                 1.0f, b1);
}

__global__ void __launch_bounds__(128)
k_gemm_ffn2(const float* __restrict__ ff1, const float* __restrict__ w2,
            const float* __restrict__ b2, float* __restrict__ ff2) {
    wmma_gemm_block<false, false>(ff1, DFF, w2, DFF, ff2, D_MODEL,
                                  DFF, blockIdx.y * 16, blockIdx.x * 64,
                                  1.0f, b2);
}

// ---------------------------------------------------------------------------
// KV-cache append via the Tensor Data Mover.
// One wave per 7-row chunk (4095 = 585 * 7). Two TENSOR_LOAD_TO_LDS pull the
// K and V tiles (28 KB each) into LDS, s_wait_tensorcnt 0, then two
// TENSOR_STORE_FROM_LDS push them to the comb_k / comb_v output regions.
// D# layout per CDNA5 ISA §8.3/8.4 (2-D tensor, groups 2/3 zero).
// ---------------------------------------------------------------------------
#define TDM_ROWS   7
#define TDM_CHUNKS 585                         // per batch
#define TDM_BYTES  (TDM_ROWS * D_MODEL * 4)    // 28672

__device__ __forceinline__ u32x4 tdm_group0(unsigned lds_off, const void* gptr) {
    const unsigned long long ga = (unsigned long long)gptr;
    u32x4 g;
    g[0] = 1u;                                              // count=1, user mode
    g[1] = lds_off;                                         // lds_addr (bytes)
    g[2] = (unsigned)(ga & 0xFFFFFFFFu);                    // global_addr[31:0]
    g[3] = (unsigned)((ga >> 32) & 0x01FFFFFFu) | (2u << 30); // addr[56:32] | type=2
    return g;
}

__device__ __forceinline__ i32x8 tdm_group1_2d(unsigned dim0, unsigned dim1,
                                               unsigned tile0, unsigned tile1,
                                               unsigned long long stride0) {
    // data_size = 3 (8-byte units); 2-D tile, no padding, no multicast
    i32x8 g;
    g[0] = (int)(3u << 16);                                        // data_size
    g[1] = (int)((dim0 & 0xFFFFu) << 16);                          // tensor_dim0 lo
    g[2] = (int)(((dim0 >> 16) & 0xFFFFu) | ((dim1 & 0xFFFFu) << 16));
    g[3] = (int)(((dim1 >> 16) & 0xFFFFu) | ((tile0 & 0xFFFFu) << 16));
    g[4] = (int)(tile1 & 0xFFFFu);                                 // tile_dim1, tile_dim2=0
    g[5] = (int)(unsigned)(stride0 & 0xFFFFFFFFu);                 // stride0[31:0]
    g[6] = (int)(unsigned)((stride0 >> 32) & 0xFFFFu);             // stride0[47:32]
    g[7] = 0;
    return g;
}

__global__ void __launch_bounds__(32)
k_copykv_tdm(const float* __restrict__ pk, const float* __restrict__ pv,
             float* __restrict__ ck, float* __restrict__ cv) {
    __shared__ float lbuf[2 * TDM_ROWS * D_MODEL];     // 56 KB staging
    ((volatile float*)lbuf)[0] = 0.0f;                 // keep LDS allocated

    const int chunk = blockIdx.x;                      // [0, 64*585)
    const int b  = chunk / TDM_CHUNKS;
    const int s0 = (chunk % TDM_CHUNKS) * TDM_ROWS;

    const float* srck = pk + ((long)b * (SEQ - 1) + s0) * D_MODEL;
    const float* srcv = pv + ((long)b * (SEQ - 1) + s0) * D_MODEL;
    float*       dstk = ck + ((long)b * SEQ + s0) * D_MODEL;
    float*       dstv = cv + ((long)b * SEQ + s0) * D_MODEL;

    const unsigned U8 = D_MODEL / 2;                   // 512 8-byte units/row
    const i32x8 g1 = tdm_group1_2d(U8, TDM_ROWS, U8, TDM_ROWS, U8);
    const i32x4 z4 = {};
#if defined(__clang_major__) && (__clang_major__ >= 23)
    const i32x8 z8 = {};
    __builtin_amdgcn_tensor_load_to_lds(tdm_group0(0, srck),          g1, z4, z4, z8, 0);
    __builtin_amdgcn_tensor_load_to_lds(tdm_group0(TDM_BYTES, srcv),  g1, z4, z4, z8, 0);
    __builtin_amdgcn_s_wait_tensorcnt(0);
    __builtin_amdgcn_tensor_store_from_lds(tdm_group0(0, dstk),         g1, z4, z4, z8, 0);
    __builtin_amdgcn_tensor_store_from_lds(tdm_group0(TDM_BYTES, dstv), g1, z4, z4, z8, 0);
#else
    __builtin_amdgcn_tensor_load_to_lds(tdm_group0(0, srck),          g1, z4, z4, 0);
    __builtin_amdgcn_tensor_load_to_lds(tdm_group0(TDM_BYTES, srcv),  g1, z4, z4, 0);
    __builtin_amdgcn_s_wait_tensorcnt(0);
    __builtin_amdgcn_tensor_store_from_lds(tdm_group0(0, dstk),         g1, z4, z4, 0);
    __builtin_amdgcn_tensor_store_from_lds(tdm_group0(TDM_BYTES, dstv), g1, z4, z4, 0);
#endif
    __builtin_amdgcn_s_wait_tensorcnt(0);
}

// append the new token x as row s=4095 of comb_k / comb_v
__global__ void __launch_bounds__(256)
k_append_x(const float4* __restrict__ x, float4* __restrict__ ck,
           float4* __restrict__ cv) {
    const int b  = blockIdx.x;
    const int d4 = threadIdx.x;                        // 0..255
    const float4 xv = x[b * (D_MODEL / 4) + d4];
    const long dst = ((long)b * SEQ + (SEQ - 1)) * (D_MODEL / 4) + d4;
    ck[dst] = xv;
    cv[dst] = xv;
}

// ---------------------------------------------------------------------------
// Softmax over S=4096 per (b, h) row, fp32, in place.
// ---------------------------------------------------------------------------
__global__ void __launch_bounds__(256)
k_softmax(float* __restrict__ scores) {
    float* p = scores + (size_t)blockIdx.x * SEQ;
    const int tid = threadIdx.x;
    __shared__ float red[256];

    float v[16], m = -3.4e38f;
    #pragma unroll
    for (int i = 0; i < 16; ++i) { v[i] = p[tid + i * 256]; m = fmaxf(m, v[i]); }
    red[tid] = m; __syncthreads();
    for (int s = 128; s > 0; s >>= 1) {
        if (tid < s) red[tid] = fmaxf(red[tid], red[tid + s]);
        __syncthreads();
    }
    m = red[0]; __syncthreads();

    float sum = 0.0f;
    #pragma unroll
    for (int i = 0; i < 16; ++i) { v[i] = __expf(v[i] - m); sum += v[i]; }
    red[tid] = sum; __syncthreads();
    for (int s = 128; s > 0; s >>= 1) {
        if (tid < s) red[tid] += red[tid + s];
        __syncthreads();
    }
    const float inv = 1.0f / red[0];
    #pragma unroll
    for (int i = 0; i < 16; ++i) p[tid + i * 256] = v[i] * inv;
}

// ---------------------------------------------------------------------------
// out[row] = LayerNorm(a[row] + b[row]) * g + beta   (row = 1024 elems)
// ---------------------------------------------------------------------------
__global__ void __launch_bounds__(256)
k_addln(const float* __restrict__ a, const float* __restrict__ b,
        const float* __restrict__ g, const float* __restrict__ be,
        float* __restrict__ out) {
    const int row = blockIdx.x, tid = threadIdx.x;
    __shared__ float red[256];
    float v[4], s = 0.0f;
    #pragma unroll
    for (int i = 0; i < 4; ++i) {
        const int c = tid + i * 256;
        v[i] = a[row * D_MODEL + c] + b[row * D_MODEL + c];
        s += v[i];
    }
    red[tid] = s; __syncthreads();
    for (int st = 128; st > 0; st >>= 1) { if (tid < st) red[tid] += red[tid + st]; __syncthreads(); }
    const float mean = red[0] * (1.0f / D_MODEL); __syncthreads();

    s = 0.0f;
    #pragma unroll
    for (int i = 0; i < 4; ++i) { const float d = v[i] - mean; s += d * d; }
    red[tid] = s; __syncthreads();
    for (int st = 128; st > 0; st >>= 1) { if (tid < st) red[tid] += red[tid + st]; __syncthreads(); }
    const float inv = rsqrtf(red[0] * (1.0f / D_MODEL) + 1e-5f);

    #pragma unroll
    for (int i = 0; i < 4; ++i) {
        const int c = tid + i * 256;
        out[row * D_MODEL + c] = (v[i] - mean) * inv * g[c] + be[c];
    }
}

// ---------------------------------------------------------------------------
extern "C" void kernel_launch(void* const* d_in, const int* in_sizes, int n_in,
                              void* d_out, int out_size, void* d_ws, size_t ws_size,
                              hipStream_t stream) {
    const float* x    = (const float*)d_in[0];
    const float* pk   = (const float*)d_in[1];
    const float* pv   = (const float*)d_in[2];
    const float* wqkv = (const float*)d_in[3];   // (3072, 1024) [N][K]
    const float* bqkv = (const float*)d_in[4];
    const float* wo   = (const float*)d_in[5];
    const float* bo   = (const float*)d_in[6];
    const float* w1   = (const float*)d_in[7];
    const float* b1   = (const float*)d_in[8];
    const float* w2   = (const float*)d_in[9];
    const float* b2   = (const float*)d_in[10];
    const float* g1   = (const float*)d_in[11];
    const float* be1  = (const float*)d_in[12];
    const float* g2   = (const float*)d_in[13];
    const float* be2  = (const float*)d_in[14];

    float* out   = (float*)d_out;                       // (64,1,1024)
    float* combk = out + BATCH * D_MODEL;               // (64,4096,1024)
    float* combv = combk + (size_t)BATCH * SEQ * D_MODEL;

    float* ws     = (float*)d_ws;
    float* q      = ws;  ws += BATCH * D_MODEL;
    float* qhat   = ws;  ws += BATCH * NHEAD * D_MODEL;
    float* scores = ws;  ws += (size_t)BATCH * NHEAD * SEQ;
    float* vbar   = ws;  ws += BATCH * NHEAD * D_MODEL;
    float* ctx    = ws;  ws += BATCH * D_MODEL;
    float* aout   = ws;  ws += BATCH * D_MODEL;
    float* hbuf   = ws;  ws += BATCH * D_MODEL;
    float* ff1    = ws;  ws += BATCH * DFF;
    float* ff2    = ws;  ws += BATCH * D_MODEL;

    // 1. stream the cache append through the Tensor Data Mover + x tail row
    k_copykv_tdm<<<dim3(BATCH * TDM_CHUNKS), 32, 0, stream>>>(pk, pv, combk, combv);
    k_append_x  <<<dim3(BATCH), 256, 0, stream>>>((const float4*)x,
                                                  (float4*)combk, (float4*)combv);

    // 2. q = x.Wq^T + bq ; qhat_h = (1/8) * Wk_h^T q_h  (bk is softmax-invariant)
    k_gemm_qproj<<<dim3(16, 4), 128, 0, stream>>>(x, wqkv, bqkv, q);
    k_gemm_qhat <<<dim3(16, 4, NHEAD), 128, 0, stream>>>(q, wqkv + (size_t)D_MODEL * D_MODEL, qhat);

    // 3. scores = qhat . comb_k^T ; softmax ; vbar = attn . comb_v
    k_gemm_scores<<<dim3(SEQ / 64, 1, BATCH), 128, 0, stream>>>(qhat, combk, scores);
    k_softmax    <<<dim3(BATCH * NHEAD), 256, 0, stream>>>(scores);
    k_gemm_vbar  <<<dim3(D_MODEL / 64, 1, BATCH), 128, 0, stream>>>(scores, combv, vbar);

    // 4. ctx_h = vbar_h . Wv_h^T + bv ; attn_out = ctx . Wo^T + bo
    k_gemm_ctx    <<<dim3(1, 4, NHEAD), 128, 0, stream>>>(vbar, wqkv + (size_t)2 * D_MODEL * D_MODEL,
                                                          bqkv + 2 * D_MODEL, ctx);
    k_gemm_outproj<<<dim3(16, 4), 128, 0, stream>>>(ctx, wo, bo, aout);

    // 5. LN1, FFN (relu fused), LN2 -> out
    k_addln    <<<dim3(BATCH), 256, 0, stream>>>(aout, x, g1, be1, hbuf);
    k_gemm_ffn1<<<dim3(DFF / 64, 4), 128, 0, stream>>>(hbuf, w1, b1, ff1);
    k_gemm_ffn2<<<dim3(16, 4), 128, 0, stream>>>(ff1, w2, b2, ff2);
    k_addln    <<<dim3(BATCH), 256, 0, stream>>>(ff2, hbuf, g2, be2, out);
}
